// SelfAttentionWide_23304492548138
// MI455X (gfx1250) — matrix operand order repaired and checked by reading, not compiled, not run on hardware
//
#include <hip/hip_runtime.h>
#include <hip/hip_bf16.h>

// ---------------------------------------------------------------------------
// SelfAttentionWide on gfx1250 (CDNA5, wave32, WMMA)
// b=16, t=512, e=256, h=8
// Pipeline:
//   1) convert x, w_q, w_k, w_v, w_u to bf16
//   2) WMMA GEMM: Q=(x Wq^T)/4 -> (b,h,t,e) bf16 ; K same ; V -> (b,h,e,t) bf16
//   3) fused attention per (b,h, 32 q-rows): S=QK^T in LDS, masked softmax,
//      P bf16 in LDS, O=PV -> (b,t,h*e) bf16   (all matmuls via v_wmma)
//   4) WMMA GEMM: y = O Wu^T + b_u -> f32 d_out
// ---------------------------------------------------------------------------

typedef __attribute__((ext_vector_type(16))) __bf16 v16bf;
typedef __attribute__((ext_vector_type(8)))  float  v8f;

#define NEG_BIG (-3.4028234663852886e38f)   // jnp.finfo(f32).min

// Load a 16-element bf16 fragment from two 8-element (16-byte) chunks.
__device__ __forceinline__ v16bf load_frag16(const __bf16* p0, const __bf16* p1) {
    union { uint4 u[2]; v16bf v; } r;
    r.u[0] = *(const uint4*)p0;
    r.u[1] = *(const uint4*)p1;
    return r.v;
}

__device__ __forceinline__ v8f wmma_bf16(v16bf a, v16bf b, v8f c) {
    return __builtin_amdgcn_wmma_f32_16x16x32_bf16(false, a, false, b,
                                                   (short)0, c, false, false);
}

// ---------------------------------------------------------------------------
// elementwise f32 -> bf16
// ---------------------------------------------------------------------------
__global__ void cvt_f32_bf16(const float* __restrict__ in,
                             __bf16* __restrict__ out, int n) {
    int i = blockIdx.x * blockDim.x + threadIdx.x;
    if (i < n) out[i] = (__bf16)in[i];
}

// ---------------------------------------------------------------------------
// Generic bf16 GEMM: C[m,n] = scale * sum_k A[m,k] * W[n,k]
//   A: M x K bf16 row-major (K contiguous)
//   W: N x K bf16 row-major (K contiguous)  => B = W^T
// Each wave computes a 64x64 tile (4x4 fragments, K stepped by 32).
// mode 0: bf16 out, (b,h,t,e) layout   (m=(bi,ti), n=(h,e))
// mode 1: bf16 out, (b,h,e,t) layout   (transposed V)
// mode 2: f32  out, row-major MxN, + bias[n]
// ---------------------------------------------------------------------------
__global__ __launch_bounds__(256)
void gemm_bf16_wmma(const __bf16* __restrict__ A, const __bf16* __restrict__ W,
                    void* __restrict__ Cout, int M, int N, int K,
                    int mode, float scale, const float* __restrict__ bias) {
    const int wave   = blockIdx.x * (blockDim.x >> 5) + (threadIdx.x >> 5);
    const int lane   = threadIdx.x & 31;
    const int r      = lane & 15;     // row/col within 16-fragment
    const int hi     = lane >> 4;     // K-half selector (A/B layouts)
    const int tilesN = N >> 6;
    const int tileM  = wave / tilesN;
    const int tileN  = wave - tileM * tilesN;
    if (tileM * 64 >= M) return;

    v8f acc[4][4] = {};

    for (int k0 = 0; k0 < K; k0 += 32) {
        v16bf af[4], bf[4];
        // A fragments: lane<16 holds K {k0..k0+7, k0+16..k0+23}; lane>=16 the +8 halves
        #pragma unroll
        for (int mf = 0; mf < 4; ++mf) {
            const __bf16* p = A + (size_t)(tileM * 64 + mf * 16 + r) * K + k0 + hi * 8;
            af[mf] = load_frag16(p, p + 16);
        }
        // B fragments: lane<16 holds K {k0..k0+7, k0+16..k0+23} of column n; lane>=16 the +8 halves
        #pragma unroll
        for (int nf = 0; nf < 4; ++nf) {
            const __bf16* p = W + (size_t)(tileN * 64 + nf * 16 + r) * K + k0 + hi * 8;
            bf[nf] = load_frag16(p, p + 16);
        }
        #pragma unroll
        for (int mf = 0; mf < 4; ++mf)
            #pragma unroll
            for (int nf = 0; nf < 4; ++nf)
                acc[mf][nf] = wmma_bf16(af[mf], bf[nf], acc[mf][nf]);
    }

    // C/D layout: VGPR i -> M = i + hi*8 ; N = lane&15
    #pragma unroll
    for (int mf = 0; mf < 4; ++mf) {
        #pragma unroll
        for (int nf = 0; nf < 4; ++nf) {
            #pragma unroll
            for (int i = 0; i < 8; ++i) {
                const int m = tileM * 64 + mf * 16 + hi * 8 + i;
                const int n = tileN * 64 + nf * 16 + r;
                const float v = acc[mf][nf][i] * scale;
                if (mode == 0) {            // (b,h,t,e)
                    const int bi = m >> 9, ti = m & 511, hh = n >> 8, e = n & 255;
                    ((__bf16*)Cout)[(((size_t)bi * 8 + hh) * 512 + ti) * 256 + e] = (__bf16)v;
                } else if (mode == 1) {     // (b,h,e,t)
                    const int bi = m >> 9, ti = m & 511, hh = n >> 8, e = n & 255;
                    ((__bf16*)Cout)[(((size_t)bi * 8 + hh) * 256 + e) * 512 + ti] = (__bf16)v;
                } else {                    // f32 row-major + bias
                    ((float*)Cout)[(size_t)m * N + n] = v + bias[n];
                }
            }
        }
    }
}

// ---------------------------------------------------------------------------
// Fused attention for one (b,h) pair and 32 query rows.
//   Q: (b,h,t,e) bf16 (pre-scaled 1/4) ; Km: same ; Vt: (b,h,e,t) bf16
//   masks: (b,t) int32 ; O: (b,t,h*e) bf16
// block = 256 threads (8 waves); grid = (b*h, t/32)
// ---------------------------------------------------------------------------
__global__ __launch_bounds__(256)
void attention_wmma(const __bf16* __restrict__ Q, const __bf16* __restrict__ Km,
                    const __bf16* __restrict__ Vt, const int* __restrict__ masks,
                    __bf16* __restrict__ O) {
    __shared__ float  S[32 * 512];    // 64 KB scores
    __shared__ __bf16 P[32 * 512];    // 32 KB probabilities
    __shared__ int    KMs[512];
    __shared__ int    QMs[32];

    const int bh = blockIdx.x;            // 0..127
    const int bi = bh >> 3;
    const int hh = bh & 7;
    const int qbase = blockIdx.y * 32;

    const int tid  = threadIdx.x;
    const int wave = tid >> 5;
    const int lane = tid & 31;
    const int r    = lane & 15;
    const int hi   = lane >> 4;

    for (int i = tid; i < 512; i += 256) KMs[i] = masks[bi * 512 + i];
    if (tid < 32) QMs[tid] = masks[bi * 512 + qbase + tid];
    __syncthreads();

    const __bf16* Qb = Q  + ((size_t)bh * 512 + qbase) * 256;
    const __bf16* Kb = Km + (size_t)bh * 512 * 256;
    const __bf16* Vb = Vt + (size_t)bh * 256 * 512;

    // ---------------- Phase 1: S = Q K^T (pre-masked) ----------------
    const int q0 = (wave >> 2) * 16;      // 0 or 16
    const int n0 = (wave & 3) * 128;      // 128-wide column slab, 8 n-tiles
    {
        v16bf afr[8];
        #pragma unroll
        for (int ks = 0; ks < 8; ++ks) {
            const __bf16* p = Qb + (size_t)(q0 + r) * 256 + ks * 32 + hi * 8;
            afr[ks] = load_frag16(p, p + 16);
        }
        for (int nt = 0; nt < 8; ++nt) {
            v8f acc = {};
            #pragma unroll
            for (int ks = 0; ks < 8; ++ks) {
                const __bf16* p = Kb + (size_t)(n0 + nt * 16 + r) * 256 + ks * 32 + hi * 8;
                acc = wmma_bf16(afr[ks], load_frag16(p, p + 16), acc);
            }
            const int col = n0 + nt * 16 + r;
            const int kmv = KMs[col];
            #pragma unroll
            for (int i = 0; i < 8; ++i) {
                const int row = q0 + hi * 8 + i;
                S[row * 512 + col] = (kmv && QMs[row]) ? acc[i] : NEG_BIG;
            }
        }
    }
    __syncthreads();

    // ---------------- Phase 2: masked softmax (8 lanes per row) ----------------
    {
        const int row = tid >> 3;
        const int seg = tid & 7;
        const float* Sr = S + row * 512 + seg * 64;
        float mx = NEG_BIG;
        for (int c = 0; c < 64; ++c) mx = fmaxf(mx, Sr[c]);
        mx = fmaxf(mx, __shfl_xor(mx, 1, 32));
        mx = fmaxf(mx, __shfl_xor(mx, 2, 32));
        mx = fmaxf(mx, __shfl_xor(mx, 4, 32));
        float sum = 0.0f;
        for (int c = 0; c < 64; ++c) sum += __expf(Sr[c] - mx);
        sum += __shfl_xor(sum, 1, 32);
        sum += __shfl_xor(sum, 2, 32);
        sum += __shfl_xor(sum, 4, 32);
        const float inv = 1.0f / sum;
        const int qmv = QMs[row];
        __bf16* Pr = P + row * 512 + seg * 64;
        const int* kmp = KMs + seg * 64;
        for (int c = 0; c < 64; ++c) {
            const float p = (qmv && kmp[c]) ? __expf(Sr[c] - mx) * inv : 0.0f;
            Pr[c] = (__bf16)p;
        }
    }
    __syncthreads();

    // ---------------- Phase 3: O = P V ----------------
    {
        const int e0 = (wave & 3) * 64;   // 64-wide embed slab, 4 n-tiles
        v8f oacc[4] = {};
        for (int ks = 0; ks < 16; ++ks) {
            const __bf16* pa = P + (size_t)(q0 + r) * 512 + ks * 32 + hi * 8;
            const v16bf a = load_frag16(pa, pa + 16);   // ds_load_b128 x2
            #pragma unroll
            for (int nt = 0; nt < 4; ++nt) {
                const __bf16* pb = Vb + (size_t)(e0 + nt * 16 + r) * 512 + ks * 32 + hi * 8;
                oacc[nt] = wmma_bf16(a, load_frag16(pb, pb + 16), oacc[nt]);
            }
        }
        #pragma unroll
        for (int nt = 0; nt < 4; ++nt) {
            const int e = e0 + nt * 16 + r;
            #pragma unroll
            for (int i = 0; i < 8; ++i) {
                const int qrow = qbase + q0 + hi * 8 + i;
                O[((size_t)bi * 512 + qrow) * 2048 + hh * 256 + e] = (__bf16)oacc[nt][i];
            }
        }
    }
}

// ---------------------------------------------------------------------------
// Host launcher
// ---------------------------------------------------------------------------
extern "C" void kernel_launch(void* const* d_in, const int* in_sizes, int n_in,
                              void* d_out, int out_size, void* d_ws, size_t ws_size,
                              hipStream_t stream) {
    (void)in_sizes; (void)n_in; (void)out_size; (void)ws_size;
    const float* x     = (const float*)d_in[0];   // (16,512,256)
    const int*   masks = (const int*)  d_in[1];   // (16,512)
    const float* w_q   = (const float*)d_in[2];   // (2048,256)
    const float* w_k   = (const float*)d_in[3];
    const float* w_v   = (const float*)d_in[4];
    const float* w_u   = (const float*)d_in[5];   // (256,2048)
    const float* b_u   = (const float*)d_in[6];   // (256,)

    const int M  = 16 * 512;          // 8192 rows of x / O
    const int F  = 2048;              // h*e
    const int E  = 256;

    // workspace layout (bf16)
    char* ws = (char*)d_ws;
    __bf16* x_bf  = (__bf16*)ws;  ws += (size_t)M * E * 2;       //  4 MiB
    __bf16* wq_bf = (__bf16*)ws;  ws += (size_t)F * E * 2;       //  1 MiB
    __bf16* wk_bf = (__bf16*)ws;  ws += (size_t)F * E * 2;
    __bf16* wv_bf = (__bf16*)ws;  ws += (size_t)F * E * 2;
    __bf16* wu_bf = (__bf16*)ws;  ws += (size_t)E * F * 2;
    __bf16* q_bf  = (__bf16*)ws;  ws += (size_t)M * F * 2;       // 32 MiB (b,h,t,e)
    __bf16* k_bf  = (__bf16*)ws;  ws += (size_t)M * F * 2;       // 32 MiB (b,h,t,e)
    __bf16* vt_bf = (__bf16*)ws;  ws += (size_t)M * F * 2;       // 32 MiB (b,h,e,t)
    __bf16* o_bf  = (__bf16*)ws;  ws += (size_t)M * F * 2;       // 32 MiB (b,t,h*e)

    // 1) convert inputs to bf16
    cvt_f32_bf16<<<(M * E + 255) / 256, 256, 0, stream>>>(x,   x_bf,  M * E);
    cvt_f32_bf16<<<(F * E + 255) / 256, 256, 0, stream>>>(w_q, wq_bf, F * E);
    cvt_f32_bf16<<<(F * E + 255) / 256, 256, 0, stream>>>(w_k, wk_bf, F * E);
    cvt_f32_bf16<<<(F * E + 255) / 256, 256, 0, stream>>>(w_v, wv_bf, F * E);
    cvt_f32_bf16<<<(E * F + 255) / 256, 256, 0, stream>>>(w_u, wu_bf, E * F);

    // 2) projections: wave-tiles = (M/64)*(N/64); 8 waves per block
    {
        const int tiles = (M / 64) * (F / 64);        // 4096
        const int blocks = tiles / 8;                 // 512
        const float s = 0.25f;                        // 1/e^0.25 per operand
        gemm_bf16_wmma<<<blocks, 256, 0, stream>>>(x_bf, wq_bf, q_bf,  M, F, E, 0, s, nullptr);
        gemm_bf16_wmma<<<blocks, 256, 0, stream>>>(x_bf, wk_bf, k_bf,  M, F, E, 0, s, nullptr);
        gemm_bf16_wmma<<<blocks, 256, 0, stream>>>(x_bf, wv_bf, vt_bf, M, F, E, 1, 1.0f, nullptr);
    }

    // 3) fused masked attention: grid (b*h, t/32)
    attention_wmma<<<dim3(16 * 8, 512 / 32), 256, 0, stream>>>(q_bf, k_bf, vt_bf, masks, o_bf);

    // 4) output projection: y = O Wu^T + b_u  (f32 out)
    {
        const int tiles = (M / 64) * (E / 64);        // 512
        const int blocks = tiles / 8;                 // 64
        gemm_bf16_wmma<<<blocks, 256, 0, stream>>>(o_bf, wu_bf, d_out, M, E, F, 2, 1.0f, b_u);
    }
}
